// resSru_39427799777304
// MI455X (gfx1250) — matrix operand order repaired and checked by reading, not compiled, not run on hardware
//
#include <hip/hip_runtime.h>
#include <hip/hip_bf16.h>

// ---------------------------------------------------------------------------
// MI455X (gfx1250): ResNet18-ish CNN + 6x bidirectional SRU, fp16-WMMA GEMMs.
// Internal activations are NHWC (coalesced implicit-GEMM loads + epilogues).
// Block tile 128x64, 8 waves, each wave 32x32 -> 4 v_wmma_f32_16x16x32_f16
// per K-step. f32->f16 conversion happens while staging tiles into LDS.
// ---------------------------------------------------------------------------

typedef __attribute__((ext_vector_type(16))) _Float16 v16h;
typedef __attribute__((ext_vector_type(8)))  float    v8f;

#define TM  128
#define TN  64
#define TK  32
#define LDH 40   // padded LDS row stride in halves

// ---- WMMA fragment loaders (CDNA5 ISA 7.12.2, wave32) ----------------------

// A fragment 16x32 (MxK): lane%16 -> M; v<4: K=8*(lane/16)+2v, v>=4: K=16+8*(lane/16)+2(v-4)
__device__ __forceinline__ v16h frag_A(const _Float16* base, int row0) {
  int lane = threadIdx.x & 31;
  int lrow = lane & 15, lhi = lane >> 4;
  const _Float16* r = base + (row0 + lrow) * LDH;
  union { v16h h; unsigned u[8]; } f;
#pragma unroll
  for (int v = 0; v < 8; ++v) {
    int k = (v < 4) ? (lhi * 8 + 2 * v) : (16 + lhi * 8 + 2 * (v - 4));
    f.u[v] = *reinterpret_cast<const unsigned*>(r + k);
  }
  return f.h;
}

// B fragment 32x16 (KxN): lane%16 -> N; v<4: K=8*(lane/16)+2v, v>=4: K=16+8*(lane/16)+2(v-4). LDS holds B as [N][K].
__device__ __forceinline__ v16h frag_B(const _Float16* base, int n0) {
  int lane = threadIdx.x & 31;
  int lrow = lane & 15, lhi = lane >> 4;
  const _Float16* r = base + (n0 + lrow) * LDH;
  union { v16h h; unsigned u[8]; } f;
#pragma unroll
  for (int v = 0; v < 8; ++v)
    f.u[v] = *reinterpret_cast<const unsigned*>(r + ((v < 4) ? (lhi * 8 + 2 * v) : (16 + lhi * 8 + 2 * (v - 4))));
  return f.h;
}

// pack 4 f32 -> 4 f16 in a 64-bit word
__device__ __forceinline__ unsigned long long pack4h(float a, float b, float c, float d) {
  union { _Float16 h[4]; unsigned long long u; } p;
  p.h[0] = (_Float16)a; p.h[1] = (_Float16)b; p.h[2] = (_Float16)c; p.h[3] = (_Float16)d;
  return p.u;
}

// ---- small utility kernels --------------------------------------------------

// conv weight OIHW f32 -> [O][KH][KW][I] f16 (K ordering = kh,kw,ci to match NHWC)
__global__ void repack_weight_kernel(const float* __restrict__ src, _Float16* __restrict__ dst,
                                     int Cout, int Cin, int KH, int KW) {
  long i = (long)blockIdx.x * blockDim.x + threadIdx.x;
  long total = (long)Cout * Cin * KH * KW;
  if (i >= total) return;
  int ci = (int)(i % Cin); long r = i / Cin;
  int kw = (int)(r % KW); r /= KW;
  int kh = (int)(r % KH); int n = (int)(r / KH);
  dst[i] = (_Float16)src[((long)(n * Cin + ci) * KH + kh) * KW + kw];
}

// RNN weight (K,N) f32 -> (N,K) f16
__global__ void transpose_f16_kernel(const float* __restrict__ W,
                                     _Float16* __restrict__ Wt, int K, int N) {
  long i = (long)blockIdx.x * blockDim.x + threadIdx.x;
  if (i >= (long)K * N) return;
  int n = (int)(i % N);
  int k = (int)(i / N);
  Wt[(long)n * K + k] = (_Float16)W[i];
}

__global__ void bn_prep_kernel(const float* g, const float* b, const float* m,
                               const float* v, float* scale, float* shift, int C) {
  int i = blockIdx.x * blockDim.x + threadIdx.x;
  if (i >= C) return;
  float s = g[i] * rsqrtf(v[i] + 1e-5f);
  scale[i] = s;
  shift[i] = b[i] - m[i] * s;
}

// ---- stem conv 7x7 s2 p3 + BN + ReLU; x is NCHW input, out is NHWC ----------

__global__ void stem_kernel(const float* __restrict__ x, const float* __restrict__ w,
                            const float* __restrict__ scale, const float* __restrict__ shift,
                            float* __restrict__ out,
                            int Bn, int Hin, int Win, int Ho, int Wo) {
  long i = (long)blockIdx.x * blockDim.x + threadIdx.x;   // NHWC out index
  long total = (long)Bn * Ho * Wo * 64;
  if (i >= total) return;
  int co = (int)(i % 64); long t = i / 64;
  int wo = (int)(t % Wo); t /= Wo;
  int ho = (int)(t % Ho); int bb = (int)(t / Ho);
  float acc = 0.f;
  for (int ci = 0; ci < 3; ++ci)
#pragma unroll
    for (int kh = 0; kh < 7; ++kh) {
      int ih = ho * 2 - 3 + kh;
      if (ih < 0 || ih >= Hin) continue;
#pragma unroll
      for (int kw = 0; kw < 7; ++kw) {
        int iw = wo * 2 - 3 + kw;
        if (iw < 0 || iw >= Win) continue;
        acc += x[((long)(bb * 3 + ci) * Hin + ih) * Win + iw] *
               w[((co * 3 + ci) * 7 + kh) * 7 + kw];
      }
    }
  float r = acc * scale[co] + shift[co];
  out[i] = fmaxf(r, 0.f);
}

// ---- maxpool 3x3 s2 p1, NHWC ------------------------------------------------

__global__ void maxpool_kernel(const float* __restrict__ in, float* __restrict__ out,
                               int Bn, int C, int Hin, int Win, int Ho, int Wo) {
  long i = (long)blockIdx.x * blockDim.x + threadIdx.x;
  long total = (long)Bn * Ho * Wo * C;
  if (i >= total) return;
  int c = (int)(i % C); long t = i / C;
  int wo = (int)(t % Wo); t /= Wo;
  int ho = (int)(t % Ho); int bb = (int)(t / Ho);
  float m = -3.4e38f;
#pragma unroll
  for (int kh = 0; kh < 3; ++kh) {
    int ih = ho * 2 - 1 + kh;
    if (ih < 0 || ih >= Hin) continue;
#pragma unroll
    for (int kw = 0; kw < 3; ++kw) {
      int iw = wo * 2 - 1 + kw;
      if (iw < 0 || iw >= Win) continue;
      m = fmaxf(m, in[((long)(bb * Hin + ih) * Win + iw) * C + c]);
    }
  }
  out[i] = m;
}

// ---- implicit-GEMM conv via WMMA (NHWC), fused BN (+residual)(+ReLU) -------
// M = B*Ho*Wo, N = Cout, K = KH*KW*Cin (kh,kw outer, ci inner).
// Requires M%128==0, N%64==0, Cin%32==0 (true at all call sites).

__global__ __launch_bounds__(256) void conv_wmma_kernel(
    const float* __restrict__ in, const _Float16* __restrict__ wgt,
    const float* __restrict__ scale, const float* __restrict__ shift,
    const float* __restrict__ resid, float* __restrict__ out,
    int Bn, int Cin, int Hin, int Win, int Cout, int Ho, int Wo,
    int KH, int KW, int sh, int sw, int ph, int pw, int relu) {
  __shared__ __align__(16) _Float16 Alds[TM * LDH];
  __shared__ __align__(16) _Float16 Blds[TN * LDH];
  const int K = Cin * KH * KW;
  const int m0 = blockIdx.y * TM;
  const int n0 = blockIdx.x * TN;
  const int tid = (int)threadIdx.x;
  const int wave = tid >> 5;
  const int wm = (wave & 3) * 32;   // 4 waves cover M=128
  const int wn = (wave >> 2) * 32;  // 2 waves cover N=64

  v8f acc[2][2] = {};

  for (int kk = 0; kk < K; kk += TK) {
    // K-chunk lies inside one (kh,kw) run since Cin % 32 == 0
    int kh = kk / (KW * Cin);
    int rem = kk - kh * (KW * Cin);
    int kw = rem / Cin;
    int ci0 = rem - kw * Cin;
    // A: 128x32 f32 quads -> f16 LDS (global_load_b128 + ds_store_b64)
#pragma unroll
    for (int i = 0; i < 4; ++i) {
      int idx4 = tid + i * 256;           // 1024 quads
      int row = idx4 >> 3, k4 = (idx4 & 7) * 4;
      int m = m0 + row;
      int wo = m % Wo; int t = m / Wo;
      int ho = t % Ho; int bb = t / Ho;
      int ih = ho * sh - ph + kh;
      int iw = wo * sw - pw + kw;
      unsigned long long pk = 0ULL;
      if (ih >= 0 && ih < Hin && iw >= 0 && iw < Win) {
        const float4 q = *reinterpret_cast<const float4*>(
            in + ((long)(bb * Hin + ih) * Win + iw) * Cin + ci0 + k4);
        pk = pack4h(q.x, q.y, q.z, q.w);
      }
      *reinterpret_cast<unsigned long long*>(&Alds[row * LDH + k4]) = pk;
    }
    // B: 64x32 f16 quads (contiguous in repacked weights)
#pragma unroll
    for (int i = 0; i < 2; ++i) {
      int idx4 = tid + i * 256;           // 512 quads
      int row = idx4 >> 3, k4 = (idx4 & 7) * 4;
      *reinterpret_cast<unsigned long long*>(&Blds[row * LDH + k4]) =
          *reinterpret_cast<const unsigned long long*>(wgt + (long)(n0 + row) * K + kk + k4);
    }
    __syncthreads();
    v16h a0 = frag_A(Alds, wm);
    v16h a1 = frag_A(Alds, wm + 16);
    v16h b0 = frag_B(Blds, wn);
    v16h b1 = frag_B(Blds, wn + 16);
    acc[0][0] = __builtin_amdgcn_wmma_f32_16x16x32_f16(false, a0, false, b0, (short)0, acc[0][0], false, false);
    acc[0][1] = __builtin_amdgcn_wmma_f32_16x16x32_f16(false, a0, false, b1, (short)0, acc[0][1], false, false);
    acc[1][0] = __builtin_amdgcn_wmma_f32_16x16x32_f16(false, a1, false, b0, (short)0, acc[1][0], false, false);
    acc[1][1] = __builtin_amdgcn_wmma_f32_16x16x32_f16(false, a1, false, b1, (short)0, acc[1][1], false, false);
    __syncthreads();
  }

  // epilogue: BN, optional residual, optional ReLU; NHWC stores (lane-contiguous)
  int lane = tid & 31, lrow = lane & 15, lhi = lane >> 4;
#pragma unroll
  for (int mi = 0; mi < 2; ++mi) {
#pragma unroll
    for (int r = 0; r < 8; ++r) {
      int m = m0 + wm + mi * 16 + r + lhi * 8;
      int wo = m % Wo; int t = m / Wo;
      int ho = t % Ho; int bb = t / Ho;
      long obase = ((long)(bb * Ho + ho) * Wo + wo) * Cout;
#pragma unroll
      for (int ni = 0; ni < 2; ++ni) {
        int n = n0 + wn + ni * 16 + lrow;
        float v = acc[mi][ni][r];
        v = v * scale[n] + shift[n];
        if (resid) v += resid[obase + n];
        if (relu) v = fmaxf(v, 0.f);
        out[obase + n] = v;
      }
    }
  }
}

// ---- plain GEMM via WMMA: C[M,N] = A[M,K] (f32) x Bt[N,K] (f16) ------------
// Requires M%128==0, N%64==0, K%32==0.

__global__ __launch_bounds__(256) void gemm_wmma_kernel(
    const float* __restrict__ A, const _Float16* __restrict__ Bt,
    float* __restrict__ C, int M, int N, int K) {
  __shared__ __align__(16) _Float16 Alds[TM * LDH];
  __shared__ __align__(16) _Float16 Blds[TN * LDH];
  const int m0 = blockIdx.y * TM;
  const int n0 = blockIdx.x * TN;
  const int tid = (int)threadIdx.x;
  const int wave = tid >> 5;
  const int wm = (wave & 3) * 32;
  const int wn = (wave >> 2) * 32;
  v8f acc[2][2] = {};
  for (int kk = 0; kk < K; kk += TK) {
#pragma unroll
    for (int i = 0; i < 4; ++i) {
      int idx4 = tid + i * 256;
      int row = idx4 >> 3, k4 = (idx4 & 7) * 4;
      const float4 q = *reinterpret_cast<const float4*>(A + (long)(m0 + row) * K + kk + k4);
      *reinterpret_cast<unsigned long long*>(&Alds[row * LDH + k4]) = pack4h(q.x, q.y, q.z, q.w);
    }
#pragma unroll
    for (int i = 0; i < 2; ++i) {
      int idx4 = tid + i * 256;
      int row = idx4 >> 3, k4 = (idx4 & 7) * 4;
      *reinterpret_cast<unsigned long long*>(&Blds[row * LDH + k4]) =
          *reinterpret_cast<const unsigned long long*>(Bt + (long)(n0 + row) * K + kk + k4);
    }
    __syncthreads();
    v16h a0 = frag_A(Alds, wm);
    v16h a1 = frag_A(Alds, wm + 16);
    v16h b0 = frag_B(Blds, wn);
    v16h b1 = frag_B(Blds, wn + 16);
    acc[0][0] = __builtin_amdgcn_wmma_f32_16x16x32_f16(false, a0, false, b0, (short)0, acc[0][0], false, false);
    acc[0][1] = __builtin_amdgcn_wmma_f32_16x16x32_f16(false, a0, false, b1, (short)0, acc[0][1], false, false);
    acc[1][0] = __builtin_amdgcn_wmma_f32_16x16x32_f16(false, a1, false, b0, (short)0, acc[1][0], false, false);
    acc[1][1] = __builtin_amdgcn_wmma_f32_16x16x32_f16(false, a1, false, b1, (short)0, acc[1][1], false, false);
    __syncthreads();
  }
  int lane = tid & 31, lrow = lane & 15, lhi = lane >> 4;
#pragma unroll
  for (int mi = 0; mi < 2; ++mi)
#pragma unroll
    for (int r = 0; r < 8; ++r) {
      int m = m0 + wm + mi * 16 + r + lhi * 8;
#pragma unroll
      for (int ni = 0; ni < 2; ++ni) {
        int n = n0 + wn + ni * 16 + lrow;
        C[(long)m * N + n] = acc[mi][ni][r];
      }
    }
}

// ---- feat NHWC (B,2,Wd,C) -> seq (L,B, C*2): s[t][b][c*2+h] = feat[b][h][t][c]

__global__ void feat_to_seq_kernel(const float* __restrict__ feat, float* __restrict__ s,
                                   int Bn, int C, int H, int Wd) {
  long i = (long)blockIdx.x * blockDim.x + threadIdx.x;   // s index
  long total = (long)Wd * Bn * C * H;
  if (i >= total) return;
  int ch2 = (int)(i % (C * H));
  int c = ch2 >> 1, h = ch2 & 1;          // H == 2
  long r = i / (C * H);
  int bb = (int)(r % Bn);
  int t = (int)(r / Bn);
  s[i] = feat[((long)(bb * H + h) * Wd + t) * C + c];
}

// ---- SRU recurrence: one thread per (b, dir, j); sequential over L ----------

__global__ void sru_scan_kernel(const float* __restrict__ u, const float* __restrict__ bias,
                                const float* __restrict__ s_in, float* __restrict__ s_out,
                                int L, int Bn, int n_out, int kch) {
  int i = blockIdx.x * blockDim.x + threadIdx.x;
  if (i >= Bn * 2 * n_out) return;
  int j = i % n_out;
  int dir = (i / n_out) & 1;
  int bb = i / (2 * n_out);
  float bf = bias[dir * n_out + j];
  float br = bias[2 * n_out + dir * n_out + j];
  int Nu = 2 * n_out * kch;
  long ubase = (long)bb * Nu + dir * n_out * kch + (long)j * kch;
  long ustride = (long)Bn * Nu;
  int So = 2 * n_out;
  long sobase = (long)bb * So + dir * n_out + j;
  float c = 0.f;
  int t = (dir == 0) ? 0 : L - 1;
  int dt = (dir == 0) ? 1 : -1;
  for (int s = 0; s < L; ++s, t += dt) {
    const float* up = u + (long)t * ustride + ubase;
    float u0 = up[0];
    float f = 1.f / (1.f + __expf(-(up[1] + bf)));
    float r = 1.f / (1.f + __expf(-(up[2] + br)));
    c = f * c + (1.f - f) * u0;
    float xp = (kch == 4) ? up[3] : s_in[(long)t * Bn * So + sobase];
    float h = r * tanhf(c) + (1.f - r) * xp;
    s_out[(long)t * Bn * So + sobase] = h;
  }
}

// ---- softmax over C=512 with (L,B,C)->(B,L,C) transpose ---------------------

__global__ __launch_bounds__(256) void softmax_kernel(const float* __restrict__ s,
                                                      float* __restrict__ out,
                                                      int L, int Bn, int C) {
  __shared__ float red[256];
  int row = blockIdx.x;            // row = b*L + t
  int bb = row / L, t = row % L;
  const float* src = s + ((long)t * Bn + bb) * C;
  float v0 = src[threadIdx.x], v1 = src[threadIdx.x + 256];
  red[threadIdx.x] = fmaxf(v0, v1);
  __syncthreads();
  for (int o = 128; o > 0; o >>= 1) {
    if (threadIdx.x < o) red[threadIdx.x] = fmaxf(red[threadIdx.x], red[threadIdx.x + o]);
    __syncthreads();
  }
  float mx = red[0];
  __syncthreads();
  float e0 = __expf(v0 - mx), e1 = __expf(v1 - mx);
  red[threadIdx.x] = e0 + e1;
  __syncthreads();
  for (int o = 128; o > 0; o >>= 1) {
    if (threadIdx.x < o) red[threadIdx.x] += red[threadIdx.x + o];
    __syncthreads();
  }
  float inv = 1.f / red[0];
  float* dst = out + (long)row * C;
  dst[threadIdx.x] = e0 * inv;
  dst[threadIdx.x + 256] = e1 * inv;
}

// ---------------------------------------------------------------------------
// Host orchestration
// ---------------------------------------------------------------------------

extern "C" void kernel_launch(void* const* d_in, const int* in_sizes, int n_in,
                              void* d_out, int out_size, void* d_ws, size_t ws_size,
                              hipStream_t stream) {
  (void)in_sizes; (void)out_size; (void)ws_size;
  // Inputs assumed flattened jax-pytree style (sorted dict keys, None dropped):
  // cnn_params{blocks, bn1, conv1}, rnn_params, x. Block i layout:
  //   s+0..3 bn1(g,b,m,v); s+4..7 bn2; s+8 conv1; s+9 conv2;
  //   [down] s+10 downW; s+11..14 down bn(g,b,m,v)
  if (n_in < 113) return;
  const int blkStart[8] = {0, 10, 20, 35, 45, 60, 70, 85};
  const int blkDown[8]  = {0, 0, 1, 0, 1, 0, 1, 0};
  const int blkCin[8]   = {64, 64, 64, 128, 128, 256, 256, 512};
  const int blkCout[8]  = {64, 64, 128, 128, 256, 256, 512, 512};
  const int blkSh[8]    = {1, 1, 2, 1, 2, 1, 2, 1};
  const int iStemBN = 95, iStemW = 99, iRnn = 100, iX = 112;
  const int B = 32, L = 256;

  auto F = [&](int i) { return (const float*)d_in[i]; };

  size_t off = 0;
  auto alloc = [&](size_t bytes) -> void* {
    off = (off + 255) & ~(size_t)255;
    void* p = (char*)d_ws + off;
    off += bytes;
    return p;
  };
  auto repack = [&](const float* src, int Co, int Ci, int KH) -> _Float16* {
    long n = (long)Co * Ci * KH * KH;
    _Float16* dst = (_Float16*)alloc((size_t)n * sizeof(_Float16));
    repack_weight_kernel<<<(unsigned)((n + 255) / 256), 256, 0, stream>>>(src, dst, Co, Ci, KH, KH);
    return dst;
  };
  auto bnprep = [&](int idx, int C) -> float* {
    float* sc = (float*)alloc((size_t)C * 2 * sizeof(float));
    bn_prep_kernel<<<(C + 255) / 256, 256, 0, stream>>>(
        F(idx), F(idx + 1), F(idx + 2), F(idx + 3), sc, sc + C, C);
    return sc;  // scale at [0,C), shift at [C,2C)
  };

  float* bnStem = bnprep(iStemBN, 64);
  _Float16 *wC1[8], *wC2[8], *wD[8];
  float *bn1[8], *bn2[8], *bnD[8];
  for (int i = 0; i < 8; ++i) {
    int s = blkStart[i], Ci = blkCin[i], Co = blkCout[i];
    bn1[i] = bnprep(s + 0, Co);
    bn2[i] = bnprep(s + 4, Co);
    wC1[i] = repack(F(s + 8), Co, Ci, 3);
    wC2[i] = repack(F(s + 9), Co, Co, 3);
    if (blkDown[i]) {
      wD[i] = repack(F(s + 10), Co, Ci, 1);
      bnD[i] = bnprep(s + 11, Co);
    } else { wD[i] = nullptr; bnD[i] = nullptr; }
  }
  _Float16* wRnn[6];
  for (int i = 0; i < 6; ++i) {
    int K = (i == 0) ? 1024 : 512;
    int N = (i == 0) ? 2048 : 1536;
    wRnn[i] = (_Float16*)alloc((size_t)K * N * sizeof(_Float16));
    transpose_f16_kernel<<<(unsigned)(((long)K * N + 255) / 256), 256, 0, stream>>>(
        F(iRnn + 2 * i), wRnn[i], K, N);
  }

  // activation buffers (NHWC)
  float* buf0 = (float*)alloc((size_t)32 * 32 * 512 * 64 * sizeof(float));  // stem out / sc scratch
  float* buf1 = (float*)alloc((size_t)8388608 * sizeof(float));
  float* buf2 = (float*)alloc((size_t)8388608 * sizeof(float));
  float* buf3 = (float*)alloc((size_t)8388608 * sizeof(float));
  float* s0   = (float*)alloc((size_t)L * B * 1024 * sizeof(float));
  float* ubuf = (float*)alloc((size_t)L * B * 2048 * sizeof(float));
  float* sA   = (float*)alloc((size_t)L * B * 512 * sizeof(float));
  float* sB   = (float*)alloc((size_t)L * B * 512 * sizeof(float));

  // stem: NCHW in -> NHWC (32,32,512,64)
  {
    long n = (long)32 * 32 * 512 * 64;
    stem_kernel<<<(unsigned)(n / 256), 256, 0, stream>>>(
        F(iX), F(iStemW), bnStem, bnStem + 64, buf0, B, 64, 1024, 32, 512);
  }
  // maxpool -> NHWC (32,16,256,64)
  {
    long n = (long)32 * 16 * 256 * 64;
    maxpool_kernel<<<(unsigned)(n / 256), 256, 0, stream>>>(buf0, buf1, B, 64, 32, 512, 16, 256);
  }

  auto conv = [&](const float* in, const _Float16* w, const float* bn, const float* resid,
                  float* out, int Cin, int Hin, int Win, int Cout, int Ho, int Wo,
                  int KH, int pad, int sh, int relu) {
    int M = B * Ho * Wo;
    dim3 grid(Cout / TN, M / TM);
    conv_wmma_kernel<<<grid, 256, 0, stream>>>(in, w, bn, bn + Cout, resid, out,
                                               B, Cin, Hin, Win, Cout, Ho, Wo,
                                               KH, KH, sh, 1, pad, pad, relu);
  };

  // 8 BasicBlocks (all NHWC)
  int H = 16, W = 256;
  float* xcur = buf1;
  float* spare = buf3;
  for (int i = 0; i < 8; ++i) {
    int Ci = blkCin[i], Co = blkCout[i], sh = blkSh[i];
    int Ho = H / sh, Wo = W;
    conv(xcur, wC1[i], bn1[i], nullptr, buf2, Ci, H, W, Co, Ho, Wo, 3, 1, sh, 1);
    const float* sc = xcur;
    if (blkDown[i]) {
      conv(xcur, wD[i], bnD[i], nullptr, buf0, Ci, H, W, Co, Ho, Wo, 1, 0, sh, 0);
      sc = buf0;
    }
    conv(buf2, wC2[i], bn2[i], sc, spare, Co, Ho, Wo, Co, Ho, Wo, 3, 1, 1, 1);
    float* t = xcur; xcur = spare; spare = t;
    H = Ho; W = Wo;
  }

  // features NHWC (32,2,256,512) -> sequence (256,32,1024)
  {
    long n = (long)B * 2 * 256 * 512;
    feat_to_seq_kernel<<<(unsigned)((n + 255) / 256), 256, 0, stream>>>(xcur, s0, B, 512, 2, 256);
  }

  // 6 SRU layers
  float* scur = s0;
  int nin = 1024;
  float* pp[2] = {sA, sB};
  int pi = 0;
  for (int i = 0; i < 6; ++i) {
    int kch = (i == 0) ? 4 : 3;
    int N = 512 * kch;  // 2048 or 1536
    dim3 g(N / TN, (L * B) / TM);
    gemm_wmma_kernel<<<g, 256, 0, stream>>>(scur, wRnn[i], ubuf, L * B, N, nin);
    float* snext = pp[pi]; pi ^= 1;
    sru_scan_kernel<<<(B * 2 * 256 + 255) / 256, 256, 0, stream>>>(
        ubuf, F(iRnn + 2 * i + 1), scur, snext, L, B, 256, kch);
    scur = snext;
    nin = 512;
  }

  // softmax + (L,B,C)->(B,L,C)
  softmax_kernel<<<B * L, 256, 0, stream>>>(scur, (float*)d_out, L, B, 512);
}
